// EmbeddingGenerator_19859928776761
// MI455X (gfx1250) — matrix operand order repaired and checked by reading, not compiled, not run on hardware
//
#include <hip/hip_runtime.h>
#include <hip/hip_bf16.h>

// ---------------------------------------------------------------------------
// Problem constants (from reference): B=128, T=256, E=256, 4E=1024, HW=64
// ---------------------------------------------------------------------------
#define BATCH 128
#define SEQT  256
#define EDIM  256
#define GDIM  1024            // 4*E
#define KPAD  288             // 256 (h) + 10 (x) + 1 (bias one-hot) + pad -> 9 K-tiles
#define TILEB 16              // batch rows per workgroup (WMMA M)
#define NWG   (BATCH / TILEB) // 8 LSTM workgroups

typedef __attribute__((ext_vector_type(16))) __bf16 v16bf;
typedef __attribute__((ext_vector_type(8)))  float  v8f;

union Frag16 {
    unsigned int  w[8];   // 8 dwords = 16 bf16
    uint4         q[2];
    v16bf         b;
};

__device__ __forceinline__ unsigned short f2bf(float f) {
    unsigned int u = __float_as_uint(f);
    u += 0x7FFFu + ((u >> 16) & 1u);   // round-to-nearest-even
    return (unsigned short)(u >> 16);
}
__device__ __forceinline__ float sigm(float x) { return 1.f / (1.f + __expf(-x)); }

// ---------------------------------------------------------------------------
// Prep: build combined, K-padded bf16 weight matrices.
//   Wc0 (1024 x 288) = [ W_hh0 (256) | W_ih0 (10) | b_ih0+b_hh0 | zeros ]
//   Wc1 (1024 x 288) = [ W_ih1 (256) | zeros (10) | b_ih1+b_hh1 | zeros ]
//   W2  (1024 x 256) = W_hh1
// With hIn = [h | x_t | 1.0 | 0...], one WMMA GEMM yields z incl. bias+x-proj.
// ---------------------------------------------------------------------------
__global__ void __launch_bounds__(256) prep_kernel(
    const float* __restrict__ Whh0f, const float* __restrict__ Wih0f,
    const float* __restrict__ Wih1f, const float* __restrict__ Whh1f,
    const float* __restrict__ bih0, const float* __restrict__ bhh0,
    const float* __restrict__ bih1, const float* __restrict__ bhh1,
    unsigned short* __restrict__ Wc0, unsigned short* __restrict__ Wc1,
    unsigned short* __restrict__ W2)
{
    int i = blockIdx.x * blockDim.x + threadIdx.x;
    if (i < GDIM * KPAD) {
        int r = i / KPAD, c = i % KPAD;
        unsigned short v0, v1;
        if (c < EDIM) {
            v0 = f2bf(Whh0f[r * EDIM + c]);
            v1 = f2bf(Wih1f[r * EDIM + c]);
        } else if (c < EDIM + 10) {
            v0 = f2bf(Wih0f[r * 10 + (c - EDIM)]);
            v1 = 0;
        } else if (c == EDIM + 10) {
            v0 = f2bf(bih0[r] + bhh0[r]);
            v1 = f2bf(bih1[r] + bhh1[r]);
        } else {
            v0 = 0; v1 = 0;
        }
        Wc0[i] = v0;
        Wc1[i] = v1;
    }
    if (i < GDIM * EDIM) W2[i] = f2bf(Whh1f[i]);
}

// ---------------------------------------------------------------------------
// Terrain branch: 3x3 SAME conv (1 -> 256 ch) + ReLU + spatial mean,
// then per-batch min/max normalization over the 256 channels.
// ---------------------------------------------------------------------------
__global__ void __launch_bounds__(256) terrain_kernel(
    const float* __restrict__ terrain, const float* __restrict__ conv_w,
    const float* __restrict__ conv_b, float* __restrict__ te_norm)
{
    __shared__ float tile[64 * 64];
    __shared__ float red[256];
    const int b = blockIdx.x, tid = threadIdx.x;

    for (int i = tid; i < 4096; i += 256) tile[i] = terrain[b * 4096 + i];

    float w[9];
    #pragma unroll
    for (int k = 0; k < 9; ++k) w[k] = conv_w[tid * 9 + k];
    const float cb = conv_b[tid];
    __syncthreads();

    float acc = 0.f;
    for (int y = 0; y < 64; ++y) {
        for (int x = 0; x < 64; ++x) {
            float s = cb;
            #pragma unroll
            for (int dy = 0; dy < 3; ++dy) {
                int yy = y + dy - 1;
                if (yy < 0 || yy > 63) continue;
                #pragma unroll
                for (int dx = 0; dx < 3; ++dx) {
                    int xx = x + dx - 1;
                    if (xx < 0 || xx > 63) continue;
                    s = fmaf(w[dy * 3 + dx], tile[yy * 64 + xx], s);
                }
            }
            acc += fmaxf(s, 0.f);
        }
    }
    const float v = acc * (1.f / 4096.f);

    red[tid] = v; __syncthreads();
    for (int off = 128; off; off >>= 1) {
        if (tid < off) red[tid] = fminf(red[tid], red[tid + off]);
        __syncthreads();
    }
    const float mn = red[0]; __syncthreads();
    red[tid] = v; __syncthreads();
    for (int off = 128; off; off >>= 1) {
        if (tid < off) red[tid] = fmaxf(red[tid], red[tid + off]);
        __syncthreads();
    }
    const float mx = red[0];
    const float denom = mx - mn;
    te_norm[b * 256 + tid] = (denom != 0.f) ? (v - mn) / denom : 0.f;
}

// ---------------------------------------------------------------------------
// One GEMM accumulation pass over K tiles:
//   acc[n] += hbuf(16 x K, bf16 LDS, row stride `stride`) @ W(row-major)^T
// A fragments per the 16-bit 16x32 layout; B fragments as 2x b128 per lane.
// ---------------------------------------------------------------------------
__device__ __forceinline__ void gemm_pass(
    v8f* acc, const unsigned short* __restrict__ W,
    const unsigned short* hbuf, int stride, int ktN,
    int wave, int fm, int fhi)
{
    #pragma unroll 2
    for (int kt = 0; kt < ktN; ++kt) {
        Frag16 A;
        #pragma unroll
        for (int j = 0; j < 8; ++j) {
            int k = kt * 32 + ((j >> 2) << 4) + (fhi << 3) + ((j & 3) << 1);
            A.w[j] = *(const unsigned int*)(&hbuf[fm * stride + k]);
        }
        #pragma unroll
        for (int n = 0; n < 4; ++n) {
            const int nB = wave * 64 + n * 16;
            const unsigned short* row = W + (nB + fm) * stride + kt * 32 + fhi * 8;
            Frag16 Bf;
            Bf.q[0] = *(const uint4*)(row);
            Bf.q[1] = *(const uint4*)(row + 16);
            acc[n] = __builtin_amdgcn_wmma_f32_16x16x32_bf16(
                false, A.b, false, Bf.b, (short)0, acc[n], false, false);
        }
    }
}

// ---------------------------------------------------------------------------
// Fused 2-layer LSTM. Grid = 8 independent batch tiles of 16 rows.
// Block = 512 threads = 16 waves; wave w owns gate columns [w*64, w*64+64).
// ---------------------------------------------------------------------------
__global__ void __launch_bounds__(512) lstm_kernel(
    const float* __restrict__ aws, const float* __restrict__ era,
    const unsigned short* __restrict__ Wc0,   // (1024,288) bf16 combined
    const unsigned short* __restrict__ Wc1,   // (1024,288) bf16 combined
    const unsigned short* __restrict__ W2,    // (1024,256) bf16
    float* __restrict__ hT1)
{
    __shared__ float          zbuf[TILEB * GDIM];   // 64 KB gate pre-activations
    __shared__ unsigned short hIn[TILEB * KPAD];    // 9 KB: [h0 | x_t | 1 | 0]
    __shared__ unsigned short hH1[TILEB * EDIM];    // 8 KB: layer1 hidden

    const int tid  = threadIdx.x;
    const int lane = tid & 31;
    const int wave = tid >> 5;
    const int m0   = blockIdx.x * TILEB;

    for (int i = tid; i < TILEB * KPAD; i += 512) hIn[i] = 0;
    for (int i = tid; i < TILEB * EDIM; i += 512) hH1[i] = 0;
    __syncthreads();
    if (tid < TILEB) hIn[tid * KPAD + EDIM + 10] = 0x3F80;   // bf16(1.0)

    // gate-phase ownership: row gm, cols {lane + 32*q} (stride-1 across lanes)
    const int gm = tid >> 5;
    float c0[8] = {0, 0, 0, 0, 0, 0, 0, 0};
    float c1[8] = {0, 0, 0, 0, 0, 0, 0, 0};

    // WMMA fragment lane decomposition
    const int fm  = lane & 15;
    const int fhi = (lane >> 4) & 1;

    const v8f vz = {0.f, 0.f, 0.f, 0.f, 0.f, 0.f, 0.f, 0.f};
    __syncthreads();

    for (int t = 0; t < SEQT; ++t) {
        // Opaque zero offset, re-produced by a volatile asm every iteration:
        // keeps the weight loads inside the time loop (no LICM hoist -> no
        // spills) WITHOUT erasing the pointers' global address space, so the
        // streams lower to global_load_b128 (LOADcnt only, no flat/DScnt
        // coupling with the LDS A-fragment loads).
        long zoff = 0;
        asm volatile("" : "+s"(zoff));
        const unsigned short* w0p = Wc0 + zoff;
        const unsigned short* w1p = Wc1 + zoff;
        const unsigned short* w2p = W2  + zoff;

        // -------- stage x_t into hIn cols [256, 266) as bf16 --------
        if (tid < 160) {
            int m = tid / 10, k = tid % 10;
            float v = (k < 6) ? aws[((m0 + m) * SEQT + t) * 6 + k]
                              : era[((m0 + m) * SEQT + t) * 4 + (k - 6)];
            hIn[m * KPAD + EDIM + k] = f2bf(v);
        }
        __syncthreads();

        // -------- layer 0: z0 = [h0|x|1] @ Wc0^T  (incl. x-proj + bias) -----
        {
            v8f acc[4];
            #pragma unroll
            for (int n = 0; n < 4; ++n) acc[n] = vz;
            gemm_pass(acc, w0p, hIn, KPAD, 9, wave, fm, fhi);
            #pragma unroll
            for (int n = 0; n < 4; ++n) {
                const int nB = wave * 64 + n * 16;
                #pragma unroll
                for (int r = 0; r < 8; ++r)
                    zbuf[(r + fhi * 8) * GDIM + nB + fm] = acc[n][r];
            }
        }
        __syncthreads();

        // -------- layer 0 gates (LDS only) --------
        #pragma unroll
        for (int q = 0; q < 8; ++q) {
            const int e = lane + 32 * q;
            const float zi = zbuf[gm * GDIM + e      ];
            const float zf = zbuf[gm * GDIM + e + 256];
            const float zg = zbuf[gm * GDIM + e + 512];
            const float zo = zbuf[gm * GDIM + e + 768];
            const float c  = sigm(zf) * c0[q] + sigm(zi) * tanhf(zg);
            c0[q] = c;
            hIn[gm * KPAD + e] = f2bf(sigm(zo) * tanhf(c));
        }
        __syncthreads();

        // -------- layer 1: z1 = [h0|x|1] @ Wc1^T + h1 @ W2^T --------
        {
            v8f acc[4];
            #pragma unroll
            for (int n = 0; n < 4; ++n) acc[n] = vz;
            gemm_pass(acc, w1p, hIn, KPAD, 9, wave, fm, fhi);  // incl. bias1
            gemm_pass(acc, w2p, hH1, EDIM, 8, wave, fm, fhi);
            #pragma unroll
            for (int n = 0; n < 4; ++n) {
                const int nB = wave * 64 + n * 16;
                #pragma unroll
                for (int r = 0; r < 8; ++r)
                    zbuf[(r + fhi * 8) * GDIM + nB + fm] = acc[n][r];
            }
        }
        __syncthreads();

        // -------- layer 1 gates --------
        #pragma unroll
        for (int q = 0; q < 8; ++q) {
            const int e = lane + 32 * q;
            const float zi = zbuf[gm * GDIM + e      ];
            const float zf = zbuf[gm * GDIM + e + 256];
            const float zg = zbuf[gm * GDIM + e + 512];
            const float zo = zbuf[gm * GDIM + e + 768];
            const float c  = sigm(zf) * c1[q] + sigm(zi) * tanhf(zg);
            c1[q] = c;
            const float h = sigm(zo) * tanhf(c);
            hH1[gm * EDIM + e] = f2bf(h);
            if (t == SEQT - 1) hT1[(m0 + gm) * EDIM + e] = h;
        }
        __syncthreads();
    }
}

// ---------------------------------------------------------------------------
// out[b, r, e] = hT1[b, e] + te_norm[b, e]   (r = 0..9)
// ---------------------------------------------------------------------------
__global__ void __launch_bounds__(256) combine_kernel(
    const float* __restrict__ hT1, const float* __restrict__ te_norm,
    float* __restrict__ out)
{
    int i = blockIdx.x * blockDim.x + threadIdx.x;
    if (i >= BATCH * 10 * EDIM) return;
    const int e = i & (EDIM - 1);
    const int b = i / (10 * EDIM);
    out[i] = hT1[b * EDIM + e] + te_norm[b * EDIM + e];
}

// ---------------------------------------------------------------------------
// Launch
// ---------------------------------------------------------------------------
extern "C" void kernel_launch(void* const* d_in, const int* in_sizes, int n_in,
                              void* d_out, int out_size, void* d_ws, size_t ws_size,
                              hipStream_t stream)
{
    (void)in_sizes; (void)n_in; (void)out_size; (void)ws_size;

    const float* aws     = (const float*)d_in[0];
    const float* era     = (const float*)d_in[1];
    const float* terrain = (const float*)d_in[2];
    const float* Wih0    = (const float*)d_in[3];
    const float* Whh0f   = (const float*)d_in[4];
    const float* bih0    = (const float*)d_in[5];
    const float* bhh0    = (const float*)d_in[6];
    const float* Wih1f   = (const float*)d_in[7];
    const float* Whh1f   = (const float*)d_in[8];
    const float* bih1    = (const float*)d_in[9];
    const float* bhh1    = (const float*)d_in[10];
    const float* conv_w  = (const float*)d_in[11];
    const float* conv_b  = (const float*)d_in[12];

    char* ws = (char*)d_ws;
    unsigned short* Wc0  = (unsigned short*)(ws);                // 1024*288*2 = 576 KB
    unsigned short* Wc1  = (unsigned short*)(ws + 589824);       // 576 KB
    unsigned short* W2   = (unsigned short*)(ws + 1179648);      // 512 KB
    float*          te   = (float*)(ws + 1703936);               // 128 KB
    float*          hT1v = (float*)(ws + 1835008);               // 128 KB
    float*          out  = (float*)d_out;

    prep_kernel<<<(GDIM * KPAD + 255) / 256, 256, 0, stream>>>(
        Whh0f, Wih0, Wih1f, Whh1f, bih0, bhh0, bih1, bhh1, Wc0, Wc1, W2);

    terrain_kernel<<<BATCH, 256, 0, stream>>>(terrain, conv_w, conv_b, te);

    lstm_kernel<<<NWG, 512, 0, stream>>>(aws, era, Wc0, Wc1, W2, hT1v);

    combine_kernel<<<(BATCH * 10 * EDIM + 255) / 256, 256, 0, stream>>>(
        hT1v, te, out);
}
